// GIN3_47141561041408
// MI455X (gfx1250) — matrix operand order repaired and checked by reading, not compiled, not run on hardware
//
#include <hip/hip_runtime.h>
#include <math.h>

typedef __attribute__((ext_vector_type(2)))  float  v2f;
typedef __attribute__((ext_vector_type(8)))  float  v8f;
typedef __attribute__((ext_vector_type(2)))  __bf16 v2bf;
typedef __attribute__((ext_vector_type(8)))  __bf16 v8bf;
typedef __attribute__((ext_vector_type(16))) __bf16 v16bf;

#define WAVES_PER_BLOCK 8

// ---------------------------------------------------------------------------
// Shared: reduce 8 per-wave partial 16x16 C tiles through LDS into H[256].
// C layout: VGPR j, lanes0-15 -> M=j, N=lane; lanes16-31 -> M=j+8, N=lane-16.
// ---------------------------------------------------------------------------
__device__ __forceinline__
void reduce_partials(const v8f& c, int wave, int lane, float* ldsC, float* H) {
  const int tid = threadIdx.x;
#pragma unroll
  for (int j = 0; j < 8; ++j) ldsC[wave * 256 + j * 32 + lane] = c[j];
  __syncthreads();
  {
    const int m  = tid >> 4;
    const int nn = tid & 15;
    const int j  = m & 7;
    const int ln = ((m >> 3) << 4) | nn;
    float s = 0.f;
#pragma unroll
    for (int w = 0; w < WAVES_PER_BLOCK; ++w) s += ldsC[w * 256 + j * 32 + ln];
    H[tid] = s;  // H[m*16 + f]
  }
  __syncthreads();
}

// ---------------------------------------------------------------------------
// fp32 aggregation tile: H = adj[16 rows] @ x via V_WMMA_F32_16X16X4_F32.
// Optionally emits a bf16 copy of the adj tiles it streams (CONVERT).
// ---------------------------------------------------------------------------
template <bool CONVERT>
__device__ __forceinline__
void agg_tile_f32(const float* __restrict__ adj, const float* __restrict__ xT,
                  __bf16* __restrict__ adjBF, int n, int rowbase,
                  float* ldsC, float* H) {
  const int tid  = threadIdx.x;
  const int wave = __builtin_amdgcn_readfirstlane(tid >> 5);  // uniform loop bounds
  const int lane = tid & 31;
  const int m16  = lane & 15;
  const int koff = (lane >> 4) << 1;  // K pair select: 0 or 2

  const float* aptr = adj + (size_t)(rowbase + m16) * n + koff;
  const float* bptr = xT  + (size_t)m16 * n + koff;
  __bf16* cptr = CONVERT ? (adjBF + (size_t)(rowbase + m16) * n + koff) : (__bf16*)0;

  v8f c = {};
  const int kspan = n / WAVES_PER_BLOCK;
  const int k0 = wave * kspan;
#pragma unroll 4
  for (int k = k0; k < k0 + kspan; k += 4) {
    v2f a, b;
    if (CONVERT) {
      a = __builtin_nontemporal_load((const v2f*)(aptr + k));  // fp32 adj dies after this pass
    } else {
      a = *(const v2f*)(aptr + k);
    }
    b = *(const v2f*)(bptr + k);
    if (CONVERT) {
      v2bf ab;
      ab.x = (__bf16)a.x;
      ab.y = (__bf16)a.y;
      *(v2bf*)(cptr + k) = ab;  // bf16 adj copy -> L2-resident for passes 2/3
    }
    c = __builtin_amdgcn_wmma_f32_16x16x4_f32(false, a, false, b,
                                              (short)0, c, false, false);
  }
  reduce_partials(c, wave, lane, ldsC, H);
}

// ---------------------------------------------------------------------------
// bf16 aggregation tile: H = adjBF[16 rows] @ x via V_WMMA_F32_16X16X32_BF16.
// A layout (16-bit 16x32): lane<16 holds K[0..7]+K[16..23]; lane>=16 holds
// K[8..15]+K[24..31] (two contiguous 16B chunks). B: lane=column, same K
// order as A (two 16B chunks of the transposed feature rows).
// ---------------------------------------------------------------------------
__device__ __forceinline__
void agg_tile_bf16(const __bf16* __restrict__ adjBF,
                   const __bf16* __restrict__ xTbf,
                   int n, int rowbase, float* ldsC, float* H) {
  const int tid  = threadIdx.x;
  const int wave = __builtin_amdgcn_readfirstlane(tid >> 5);
  const int lane = tid & 31;
  const int m16  = lane & 15;
  const int half = lane >> 4;

  const __bf16* aptr = adjBF + (size_t)(rowbase + m16) * n + half * 8;
  const __bf16* bptr = xTbf + (size_t)m16 * n + half * 8;

  v8f c = {};
  const int kspan = n / WAVES_PER_BLOCK;
  const int k0 = wave * kspan;
#pragma unroll 4
  for (int k = k0; k < k0 + kspan; k += 32) {
    v8bf a0 = *(const v8bf*)(aptr + k);
    v8bf a1 = *(const v8bf*)(aptr + k + 16);
    v8bf b0 = *(const v8bf*)(bptr + k);
    v8bf b1 = *(const v8bf*)(bptr + k + 16);
    v16bf A = __builtin_shufflevector(a0, a1, 0, 1, 2, 3, 4, 5, 6, 7,
                                      8, 9, 10, 11, 12, 13, 14, 15);
    v16bf B = __builtin_shufflevector(b0, b1, 0, 1, 2, 3, 4, 5, 6, 7,
                                      8, 9, 10, 11, 12, 13, 14, 15);
    c = __builtin_amdgcn_wmma_f32_16x16x32_bf16(false, A, false, B,
                                                (short)0, c, false, false);
  }
  reduce_partials(c, wave, lane, ldsC, H);
}

// ---------------------------------------------------------------------------
// MLP epilogues
// ---------------------------------------------------------------------------
template <int DIN, typename OUT_T>
__device__ __forceinline__
void layer_epilogue(const float* H, const float* W1, const float* b1,
                    const float* W2, const float* b2,
                    OUT_T* __restrict__ outT, int n, int rowbase) {
  const int tid = threadIdx.x;
  if (tid < 16) {
    const int m = tid;
    float h1[8];
#pragma unroll
    for (int j = 0; j < 8; ++j) {
      float s = b1[j];
#pragma unroll
      for (int f = 0; f < DIN; ++f) s += H[m * 16 + f] * W1[f * 8 + j];
      h1[j] = fmaxf(s, 0.f);
    }
#pragma unroll
    for (int j = 0; j < 8; ++j) {
      float s = b2[j];
#pragma unroll
      for (int f = 0; f < 8; ++f) s += h1[f] * W2[f * 8 + j];
      outT[(size_t)j * n + rowbase + m] = (OUT_T)fmaxf(s, 0.f);  // transposed
    }
  }
}

__device__ __forceinline__
void heads_epilogue(const float* H,
                    const float* W1p, const float* b1p, const float* W2p,
                    const float* b2p, const float* W1v, const float* b1v,
                    const float* W2v, const float* b2v,
                    float* pol_logit, float* val_raw, int rowbase) {
  const int tid = threadIdx.x;
  if (tid < 16) {
    float sp = b1p[0], sv = b1v[0];
#pragma unroll
    for (int f = 0; f < 8; ++f) {
      const float h = H[tid * 16 + f];
      sp += h * W1p[f];
      sv += h * W1v[f];
    }
    sp = fmaxf(sp, 0.f) * W2p[0] + b2p[0];
    sv = fmaxf(sv, 0.f) * W2v[0] + b2v[0];
    pol_logit[rowbase + tid] = sp;
    val_raw[rowbase + tid]   = sv;
  }
}

// ---------------------------------------------------------------------------
// Kernels: bf16 path
// ---------------------------------------------------------------------------
__global__ __launch_bounds__(256)
void gin_pass1_conv(const float* __restrict__ adj, const float* __restrict__ xT,
                    __bf16* __restrict__ adjBF,
                    const float* __restrict__ W1, const float* __restrict__ b1,
                    const float* __restrict__ W2, const float* __restrict__ b2,
                    __bf16* __restrict__ outTbf, int n) {
  __shared__ float ldsC[WAVES_PER_BLOCK * 256];
  __shared__ float H[256];
  const int rowbase = blockIdx.x * 16;
  agg_tile_f32<true>(adj, xT, adjBF, n, rowbase, ldsC, H);
  layer_epilogue<2, __bf16>(H, W1, b1, W2, b2, outTbf, n, rowbase);
}

__global__ __launch_bounds__(256)
void gin_layer_bf16(const __bf16* __restrict__ adjBF,
                    const __bf16* __restrict__ xTbf,
                    const float* __restrict__ W1, const float* __restrict__ b1,
                    const float* __restrict__ W2, const float* __restrict__ b2,
                    __bf16* __restrict__ outTbf, int n) {
  __shared__ float ldsC[WAVES_PER_BLOCK * 256];
  __shared__ float H[256];
  const int rowbase = blockIdx.x * 16;
  agg_tile_bf16(adjBF, xTbf, n, rowbase, ldsC, H);
  layer_epilogue<8, __bf16>(H, W1, b1, W2, b2, outTbf, n, rowbase);
}

__global__ __launch_bounds__(256)
void gin_heads_bf16(const __bf16* __restrict__ adjBF,
                    const __bf16* __restrict__ xTbf,
                    const float* __restrict__ W1p, const float* __restrict__ b1p,
                    const float* __restrict__ W2p, const float* __restrict__ b2p,
                    const float* __restrict__ W1v, const float* __restrict__ b1v,
                    const float* __restrict__ W2v, const float* __restrict__ b2v,
                    float* __restrict__ pol_logit, float* __restrict__ val_raw,
                    int n) {
  __shared__ float ldsC[WAVES_PER_BLOCK * 256];
  __shared__ float H[256];
  const int rowbase = blockIdx.x * 16;
  agg_tile_bf16(adjBF, xTbf, n, rowbase, ldsC, H);
  heads_epilogue(H, W1p, b1p, W2p, b2p, W1v, b1v, W2v, b2v,
                 pol_logit, val_raw, rowbase);
}

// ---------------------------------------------------------------------------
// Kernels: fp32 fallback path (if workspace too small for bf16 adj copy)
// ---------------------------------------------------------------------------
template <int DIN>
__global__ __launch_bounds__(256)
void gin_layer_f32(const float* __restrict__ adj, const float* __restrict__ xT,
                   const float* __restrict__ W1, const float* __restrict__ b1,
                   const float* __restrict__ W2, const float* __restrict__ b2,
                   float* __restrict__ outT, int n) {
  __shared__ float ldsC[WAVES_PER_BLOCK * 256];
  __shared__ float H[256];
  const int rowbase = blockIdx.x * 16;
  agg_tile_f32<false>(adj, xT, (__bf16*)0, n, rowbase, ldsC, H);
  layer_epilogue<DIN, float>(H, W1, b1, W2, b2, outT, n, rowbase);
}

__global__ __launch_bounds__(256)
void gin_heads_f32(const float* __restrict__ adj, const float* __restrict__ xT,
                   const float* __restrict__ W1p, const float* __restrict__ b1p,
                   const float* __restrict__ W2p, const float* __restrict__ b2p,
                   const float* __restrict__ W1v, const float* __restrict__ b1v,
                   const float* __restrict__ W2v, const float* __restrict__ b2v,
                   float* __restrict__ pol_logit, float* __restrict__ val_raw,
                   int n) {
  __shared__ float ldsC[WAVES_PER_BLOCK * 256];
  __shared__ float H[256];
  const int rowbase = blockIdx.x * 16;
  agg_tile_f32<false>(adj, xT, (__bf16*)0, n, rowbase, ldsC, H);
  heads_epilogue(H, W1p, b1p, W2p, b2p, W1v, b1v, W2v, b2v,
                 pol_logit, val_raw, rowbase);
}

// ---------------------------------------------------------------------------
// Init kernels
// ---------------------------------------------------------------------------
__global__ __launch_bounds__(256)
void gin_init_bf16(const float* __restrict__ dfp, const float* __restrict__ dts,
                   float* __restrict__ x0T, __bf16* __restrict__ c1Tbf,
                   __bf16* __restrict__ c2Tbf, int n) {
  int i = blockIdx.x * blockDim.x + threadIdx.x;
  if (i < n) {
    x0T[i]     = dfp[i];
    x0T[n + i] = dts[i];
#pragma unroll
    for (int r = 2; r < 16; ++r) x0T[(size_t)r * n + i] = 0.f;
#pragma unroll
    for (int r = 8; r < 16; ++r) {
      c1Tbf[(size_t)r * n + i] = (__bf16)0.f;
      c2Tbf[(size_t)r * n + i] = (__bf16)0.f;
    }
  }
}

__global__ __launch_bounds__(256)
void gin_init_f32(const float* __restrict__ dfp, const float* __restrict__ dts,
                  float* __restrict__ x0T, float* __restrict__ c1T,
                  float* __restrict__ c2T, int n) {
  int i = blockIdx.x * blockDim.x + threadIdx.x;
  if (i < n) {
    x0T[i]     = dfp[i];
    x0T[n + i] = dts[i];
#pragma unroll
    for (int r = 2; r < 16; ++r) x0T[(size_t)r * n + i] = 0.f;
#pragma unroll
    for (int r = 8; r < 16; ++r) {
      c1T[(size_t)r * n + i] = 0.f;
      c2T[(size_t)r * n + i] = 0.f;
    }
  }
}

// ---------------------------------------------------------------------------
// Finalize: softmax(policy) -> out[0:N]; (val-mean)/std -> out[N:2N].
// ---------------------------------------------------------------------------
__global__ __launch_bounds__(1024)
void gin_finalize(const float* __restrict__ pol, const float* __restrict__ val,
                  float* __restrict__ out, int n) {
  __shared__ float red[1024];
  const int tid = threadIdx.x;

  float m = -3.402823466e38f;
  for (int i = tid; i < n; i += 1024) m = fmaxf(m, pol[i]);
  red[tid] = m; __syncthreads();
  for (int s = 512; s > 0; s >>= 1) {
    if (tid < s) red[tid] = fmaxf(red[tid], red[tid + s]);
    __syncthreads();
  }
  const float pmax = red[0]; __syncthreads();

  float sum = 0.f;
  for (int i = tid; i < n; i += 1024) sum += expf(pol[i] - pmax);
  red[tid] = sum; __syncthreads();
  for (int s = 512; s > 0; s >>= 1) {
    if (tid < s) red[tid] += red[tid + s];
    __syncthreads();
  }
  const float inv = 1.f / red[0]; __syncthreads();
  for (int i = tid; i < n; i += 1024) out[i] = expf(pol[i] - pmax) * inv;

  float vs = 0.f;
  for (int i = tid; i < n; i += 1024) vs += val[i];
  red[tid] = vs; __syncthreads();
  for (int s = 512; s > 0; s >>= 1) {
    if (tid < s) red[tid] += red[tid + s];
    __syncthreads();
  }
  const float mean = red[0] / (float)n; __syncthreads();

  float vv = 0.f;
  for (int i = tid; i < n; i += 1024) { float d = val[i] - mean; vv += d * d; }
  red[tid] = vv; __syncthreads();
  for (int s = 512; s > 0; s >>= 1) {
    if (tid < s) red[tid] += red[tid + s];
    __syncthreads();
  }
  const float istd = rsqrtf(red[0] / (float)n + 1e-10f);
  for (int i = tid; i < n; i += 1024) out[n + i] = (val[i] - mean) * istd;
}

// ---------------------------------------------------------------------------
extern "C" void kernel_launch(void* const* d_in, const int* in_sizes, int n_in,
                              void* d_out, int out_size, void* d_ws, size_t ws_size,
                              hipStream_t stream) {
  const int n = in_sizes[1];  // N = 8192

  const float* adj = (const float*)d_in[0];
  const float* dfp = (const float*)d_in[1];
  const float* dts = (const float*)d_in[2];
  const float* W1_l0 = (const float*)d_in[3];
  const float* b1_l0 = (const float*)d_in[4];
  const float* W2_l0 = (const float*)d_in[5];
  const float* b2_l0 = (const float*)d_in[6];
  const float* W1_l1 = (const float*)d_in[7];
  const float* b1_l1 = (const float*)d_in[8];
  const float* W2_l1 = (const float*)d_in[9];
  const float* b2_l1 = (const float*)d_in[10];
  const float* W1_p  = (const float*)d_in[11];
  const float* b1_p  = (const float*)d_in[12];
  const float* W2_p  = (const float*)d_in[13];
  const float* b2_p  = (const float*)d_in[14];
  const float* W1_v  = (const float*)d_in[15];
  const float* b1_v  = (const float*)d_in[16];
  const float* W2_v  = (const float*)d_in[17];
  const float* b2_v  = (const float*)d_in[18];

  float* ws = (float*)d_ws;
  // Common small buffers
  float* x0T  = ws;                          // [16][N] fp32
  float* c1T  = ws + (size_t)16 * n;         // fp32 fallback
  float* c2T  = ws + (size_t)32 * n;         // fp32 fallback
  float* plog = ws + (size_t)48 * n;         // [N]
  float* vraw = ws + (size_t)49 * n;         // [N]
  // bf16-path buffers
  __bf16* c1Tbf = (__bf16*)(ws + (size_t)50 * n);  // [16][N] bf16
  __bf16* c2Tbf = (__bf16*)(ws + (size_t)58 * n);  // [16][N] bf16
  __bf16* adjBF = (__bf16*)(ws + (size_t)66 * n);  // [N][N] bf16 (~128 MB)

  const size_t need_bf16 =
      (size_t)66 * n * sizeof(float) + (size_t)n * (size_t)n * sizeof(unsigned short);

  if (ws_size >= need_bf16) {
    // Fast path: adj converted to bf16 in pass 1; passes 2/3 read the
    // 128 MB bf16 copy, which fits in the 192 MB L2.
    gin_init_bf16<<<(n + 255) / 256, 256, 0, stream>>>(dfp, dts, x0T,
                                                       c1Tbf, c2Tbf, n);
    gin_pass1_conv<<<n / 16, 256, 0, stream>>>(adj, x0T, adjBF, W1_l0, b1_l0,
                                               W2_l0, b2_l0, c1Tbf, n);
    gin_layer_bf16<<<n / 16, 256, 0, stream>>>(adjBF, c1Tbf, W1_l1, b1_l1,
                                               W2_l1, b2_l1, c2Tbf, n);
    gin_heads_bf16<<<n / 16, 256, 0, stream>>>(adjBF, c2Tbf, W1_p, b1_p, W2_p,
                                               b2_p, W1_v, b1_v, W2_v, b2_v,
                                               plog, vraw, n);
  } else {
    // Fallback: three fp32 passes over adj.
    gin_init_f32<<<(n + 255) / 256, 256, 0, stream>>>(dfp, dts, x0T, c1T, c2T, n);
    gin_layer_f32<2><<<n / 16, 256, 0, stream>>>(adj, x0T, W1_l0, b1_l0, W2_l0,
                                                 b2_l0, c1T, n);
    gin_layer_f32<8><<<n / 16, 256, 0, stream>>>(adj, c1T, W1_l1, b1_l1, W2_l1,
                                                 b2_l1, c2T, n);
    gin_heads_f32<<<n / 16, 256, 0, stream>>>(adj, c2T, W1_p, b1_p, W2_p, b2_p,
                                              W1_v, b1_v, W2_v, b2_v,
                                              plog, vraw, n);
  }

  gin_finalize<<<1, 1024, 0, stream>>>(plog, vraw, (float*)d_out, n);
}
